// MixtureOfAdaptors_240518168737
// MI455X (gfx1250) — matrix operand order repaired and checked by reading, not compiled, not run on hardware
//
#include <hip/hip_runtime.h>

typedef __attribute__((ext_vector_type(16))) _Float16 v16h;
typedef __attribute__((ext_vector_type(8)))  _Float16 v8h;
typedef __attribute__((ext_vector_type(4)))  _Float16 v4h;
typedef __attribute__((ext_vector_type(8)))  float    v8f;
typedef __attribute__((ext_vector_type(4)))  float    v4f;

#define HIDDEN 1024
#define BATCH  65536
#define BM 128
#define BN 128
#define BK 32
#define XS 40            // padded LDS row stride in halfs (80B: 16B-aligned, conflict-free)
#define KSTEPS (HIDDEN / BK)

__device__ __forceinline__ v16h frag16(const _Float16* p0, const _Float16* p1) {
    v8h a = *(const v8h*)p0;
    v8h b = *(const v8h*)p1;
    return __builtin_shufflevector(a, b, 0,1,2,3,4,5,6,7,8,9,10,11,12,13,14,15);
}

__global__ __launch_bounds__(256, 2)
void MixtureOfAdaptors_gemm_kernel(const float* __restrict__ x,
                                   const float* __restrict__ W0,
                                   const float* __restrict__ b0,
                                   float* __restrict__ out)
{
    __shared__ _Float16 sXh[BM * XS];
    __shared__ _Float16 sXl[BM * XS];
    __shared__ _Float16 sWh[BN * XS];
    __shared__ _Float16 sWl[BN * XS];

    const int tid   = threadIdx.x;
    const int lane  = tid & 31;
    const int wave  = tid >> 5;
    const int wm    = wave & 1;      // 0..1 : which 64-row slab
    const int wn    = wave >> 1;     // 0..3 : which 32-col slab (2 N-tiles)
    const int lhalf = lane >> 4;     // lane 0-15 vs 16-31 (K-half: k = 16*lhalf + e in this code)
    const int l16   = lane & 15;

    const int row0 = blockIdx.y * BM;
    const int col0 = blockIdx.x * BN;

    v8f zero = {};
    v8f acc[4][2];
    #pragma unroll
    for (int t = 0; t < 4; ++t)
        #pragma unroll
        for (int u = 0; u < 2; ++u) acc[t][u] = zero;

    for (int kt = 0; kt < KSTEPS; ++kt) {
        const int kb = kt * BK;

        // ---- stage 1: coalesced global loads (b128) into registers ----
        v4f xv[4], wv[4];
        #pragma unroll
        for (int j = 0; j < 4; ++j) {
            int i  = tid + 256 * j;          // float4 index within 128x32 tile
            int r  = i >> 3;                 // 0..127
            int c4 = i & 7;                  // 0..7
            xv[j] = *(const v4f*)(x  + (size_t)(row0 + r) * HIDDEN + kb + c4 * 4);
            wv[j] = *(const v4f*)(W0 + (size_t)(col0 + r) * HIDDEN + kb + c4 * 4);
        }

        __syncthreads();  // previous iteration's fragment reads complete

        // ---- stage 2: f32 -> (f16 hi, f16 lo) split, store planes to LDS ----
        #pragma unroll
        for (int j = 0; j < 4; ++j) {
            int i  = tid + 256 * j;
            int r  = i >> 3;
            int c4 = i & 7;
            v4h xh, xl, wh, wl;
            #pragma unroll
            for (int e = 0; e < 4; ++e) {
                float v = xv[j][e];
                _Float16 hh = (_Float16)v;
                xh[e] = hh;
                xl[e] = (_Float16)(v - (float)hh);
                float w = wv[j][e];
                _Float16 wh16 = (_Float16)w;
                wh[e] = wh16;
                wl[e] = (_Float16)(w - (float)wh16);
            }
            *(v4h*)(sXh + r * XS + c4 * 4) = xh;
            *(v4h*)(sXl + r * XS + c4 * 4) = xl;
            *(v4h*)(sWh + r * XS + c4 * 4) = wh;
            *(v4h*)(sWl + r * XS + c4 * 4) = wl;
        }

        __syncthreads();  // tiles visible

        // ---- stage 3: fragments (contiguous k = 16*lhalf + e, same for A and B) + split WMMAs ----
        // B 32x16: lane half selects K 0-15 / 16-31, 16 contiguous halfs.
        v16h bh[2], bl[2];
        #pragma unroll
        for (int u = 0; u < 2; ++u) {
            const _Float16* pbh = sWh + (wn * 32 + u * 16 + l16) * XS + lhalf * 16;
            const _Float16* pbl = sWl + (wn * 32 + u * 16 + l16) * XS + lhalf * 16;
            bh[u] = frag16(pbh, pbh + 8);
            bl[u] = frag16(pbl, pbl + 8);
        }

        #pragma unroll
        for (int t = 0; t < 4; ++t) {
            const int m = wm * 64 + t * 16 + l16;
            // A 16x32: elems 0-15 at k = lhalf*16 + e (matches B)
            const _Float16* pah = sXh + m * XS + lhalf * 16;
            const _Float16* pal = sXl + m * XS + lhalf * 16;
            v16h ah = frag16(pah, pah + 8);
            v16h al = frag16(pal, pal + 8);
            #pragma unroll
            for (int u = 0; u < 2; ++u) {
                acc[t][u] = __builtin_amdgcn_wmma_f32_16x16x32_f16(false, ah, false, bh[u],
                                                                   (short)0, acc[t][u], false, false);
                acc[t][u] = __builtin_amdgcn_wmma_f32_16x16x32_f16(false, ah, false, bl[u],
                                                                   (short)0, acc[t][u], false, false);
                acc[t][u] = __builtin_amdgcn_wmma_f32_16x16x32_f16(false, al, false, bh[u],
                                                                   (short)0, acc[t][u], false, false);
            }
        }
    }

    // ---- epilogue: bias + store (C/D layout: VGPR j -> M = j + 8*lhalf, N = l16) ----
    #pragma unroll
    for (int u = 0; u < 2; ++u) {
        const int ocol = col0 + wn * 32 + u * 16 + l16;
        const float bias = b0[ocol];
        #pragma unroll
        for (int t = 0; t < 4; ++t) {
            #pragma unroll
            for (int j = 0; j < 8; ++j) {
                int orow = row0 + wm * 64 + t * 16 + lhalf * 8 + j;
                out[(size_t)orow * HIDDEN + ocol] = acc[t][u][j] + bias;
            }
        }
    }
}

extern "C" void kernel_launch(void* const* d_in, const int* in_sizes, int n_in,
                              void* d_out, int out_size, void* d_ws, size_t ws_size,
                              hipStream_t stream) {
    (void)in_sizes; (void)n_in; (void)out_size; (void)d_ws; (void)ws_size;
    const float* x  = (const float*)d_in[0];
    // d_in[1] = routing_vectors (dead in the reference)
    const float* W  = (const float*)d_in[2];   // adaptor 0 = first 1024x1024 block
    const float* b  = (const float*)d_in[3];   // adaptor 0 = first 1024
    float* out = (float*)d_out;

    dim3 grid(HIDDEN / BN, BATCH / BM);  // col-tiles fastest: x-panel reuse hits L2
    dim3 block(256);
    hipLaunchKernelGGL(MixtureOfAdaptors_gemm_kernel, grid, block, 0, stream,
                       x, W, b, out);
}
